// LSTMLM_21380347200215
// MI455X (gfx1250) — compile-verified, matrix operand order checked by reading, not run on hardware
//
#include <hip/hip_runtime.h>
#include <hip/hip_bf16.h>

// ---------------- problem constants ----------------
constexpr int VOCAB = 4096, EMB = 256, HID = 512;
constexpr int BB = 64, TT = 1024;
constexpr int G4 = 4 * HID;                   // 2048 gate dims
constexpr long long NBT = (long long)BB * TT; // 65536 rows

constexpr int LSTM_BLOCKS = 16;               // 16 blocks x 8 waves = 128 waves = 4x32 patches

// ---------------- workspace layout (bytes) ----------------
constexpr size_t SZ_XPROJ = (size_t)NBT * G4 * 4;      // 512 MB f32, laid out [T][B][G4]
constexpr size_t SZ_HSB   = (size_t)NBT * HID * 2;     // 64 MB bf16 hidden states [B][T][H]
constexpr size_t SZ_EMBB  = (size_t)VOCAB * EMB * 2;   // 2 MB
constexpr size_t SZ_WIHB  = (size_t)G4 * EMB * 2;      // 1 MB
constexpr size_t SZ_WHHB  = (size_t)G4 * HID * 2;      // 2 MB
constexpr size_t SZ_WPRB  = (size_t)VOCAB * HID * 2;   // 4 MB
constexpr size_t SZ_HB    = (size_t)2 * BB * HID * 2;  // 128 KB (double-buffered h, bf16)
constexpr size_t SZ_BIAS  = (size_t)G4 * 4;            // 8 KB

constexpr size_t O_XPROJ = 0;
constexpr size_t O_HSB   = O_XPROJ + SZ_XPROJ;
constexpr size_t O_EMBB  = O_HSB   + SZ_HSB;
constexpr size_t O_WIHB  = O_EMBB  + SZ_EMBB;
constexpr size_t O_WHHB  = O_WIHB  + SZ_WIHB;
constexpr size_t O_WPRB  = O_WHHB  + SZ_WHHB;
constexpr size_t O_HB    = O_WPRB  + SZ_WPRB;
constexpr size_t O_BIAS  = O_HB    + SZ_HB;
constexpr size_t O_BAR   = O_BIAS  + SZ_BIAS;

// ---------------- WMMA fragment types (wave32, gfx1250) ----------------
typedef __attribute__((ext_vector_type(16))) __bf16 v16bf;
typedef __attribute__((ext_vector_type(8)))  float  v8f;

// 16-bit A/B fragment, 16x32 tile, ISA 7.12.2 layout:
// lane<16 : row M=lane,    K = {0..7} u {16..23}
// lane>=16: row M=lane-16, K = {8..15} u {24..31}
// p points at (row_base + k0); hk = lane>>4
__device__ __forceinline__ v16bf load_frag16(const __bf16* __restrict__ p, int hk) {
    union { v16bf v; float4 q[2]; } u;
    u.q[0] = *(const float4*)(p + (hk << 3));
    u.q[1] = *(const float4*)(p + 16 + (hk << 3));
    return u.v;
}

__device__ __forceinline__ v8f wmma_bf16(v16bf a, v16bf b, v8f c) {
    return __builtin_amdgcn_wmma_f32_16x16x32_bf16(false, a, false, b, (short)0, c, false, false);
}

__device__ __forceinline__ float fast_sigmoid(float x) {
    return 1.0f / (1.0f + __expf(-x));
}
__device__ __forceinline__ float fast_tanh(float x) {
    float e = __expf(-2.0f * x);
    return (1.0f - e) / (1.0f + e);
}

// Device-wide barrier: monotonically increasing arrival counter (one arrival
// per block per barrier); target = LSTM_BLOCKS * (#barriers so far).
__device__ __forceinline__ void grid_sync(unsigned* bar, unsigned target) {
    __syncthreads();
    if (threadIdx.x == 0) {
        __threadfence();
        __hip_atomic_fetch_add(bar, 1u, __ATOMIC_RELEASE, __HIP_MEMORY_SCOPE_AGENT);
        while (__hip_atomic_load(bar, __ATOMIC_ACQUIRE, __HIP_MEMORY_SCOPE_AGENT) < target) {
            __builtin_amdgcn_s_sleep(2);
        }
        __threadfence();
    }
    __syncthreads();
}

// ---------------- kernel 1: convert weights to bf16, init state ----------------
__global__ void k_init(const float* __restrict__ emb,  const float* __restrict__ wih,
                       const float* __restrict__ whh,  const float* __restrict__ wproj,
                       const float* __restrict__ bih,  const float* __restrict__ bhh,
                       __bf16* __restrict__ embb, __bf16* __restrict__ wihb,
                       __bf16* __restrict__ whhb, __bf16* __restrict__ wprojb,
                       float* __restrict__ bias, __bf16* __restrict__ hb,
                       unsigned* __restrict__ bar) {
    size_t i = (size_t)blockIdx.x * blockDim.x + threadIdx.x;
    size_t st = (size_t)gridDim.x * blockDim.x;
    for (size_t j = i; j < (size_t)VOCAB * EMB; j += st) embb[j]   = (__bf16)emb[j];
    for (size_t j = i; j < (size_t)G4 * EMB;    j += st) wihb[j]   = (__bf16)wih[j];
    for (size_t j = i; j < (size_t)G4 * HID;    j += st) whhb[j]   = (__bf16)whh[j];
    for (size_t j = i; j < (size_t)VOCAB * HID; j += st) wprojb[j] = (__bf16)wproj[j];
    for (size_t j = i; j < (size_t)G4;          j += st) bias[j]   = bih[j] + bhh[j];
    for (size_t j = i; j < (size_t)2 * BB * HID; j += st) hb[j] = (__bf16)0.0f;
    if (i == 0) *bar = 0u;
}

// ---------------- kernel 2: embed + input projection ----------------
// xproj[t][b][g] = sum_e emb[x[b,t],e] * W_ih[g,e] + (b_ih+b_hh)[g]
// 32x64 output strip per wave (2 A frags x 4 B frags, 8 WMMA / 12 loads per k-step)
// 2048 row strips x 32 col strips = 65536 waves = 8192 blocks
__global__ void k_xproj(const int* __restrict__ xs, const __bf16* __restrict__ embb,
                        const __bf16* __restrict__ wihb, const float* __restrict__ bias,
                        float* __restrict__ xproj) {
    int wave = threadIdx.x >> 5, lane = threadIdx.x & 31;
    int w = blockIdx.x * 8 + wave;
    int tn = w & 31, tm = w >> 5;
    int row0 = tm << 5, col0 = tn << 6;
    int nm = lane & 15, hk = lane >> 4;

    int tok0 = xs[row0 + nm];
    int tok1 = xs[row0 + 16 + nm];
    const __bf16* a0 = embb + (size_t)tok0 * EMB;
    const __bf16* a1 = embb + (size_t)tok1 * EMB;
    const __bf16* bp[4];
    #pragma unroll
    for (int ni = 0; ni < 4; ++ni) bp[ni] = wihb + (size_t)(col0 + ni * 16 + nm) * EMB;

    union Acc { v8f v; float e[8]; } acc[2][4];
    #pragma unroll
    for (int ni = 0; ni < 4; ++ni) {
        float bv = bias[col0 + ni * 16 + nm];
        #pragma unroll
        for (int mi = 0; mi < 2; ++mi)
            #pragma unroll
            for (int r = 0; r < 8; ++r) acc[mi][ni].e[r] = bv;
    }

    for (int k0 = 0; k0 < EMB; k0 += 32) {
        v16bf af0 = load_frag16(a0 + k0, hk);
        v16bf af1 = load_frag16(a1 + k0, hk);
        #pragma unroll
        for (int ni = 0; ni < 4; ++ni) {
            v16bf bf = load_frag16(bp[ni] + k0, hk);
            acc[0][ni].v = wmma_bf16(af0, bf, acc[0][ni].v);
            acc[1][ni].v = wmma_bf16(af1, bf, acc[1][ni].v);
        }
    }

    // row = b*T + t (strip is within one b since T % 32 == 0); write [T][B][G4]
    #pragma unroll
    for (int mi = 0; mi < 2; ++mi)
        #pragma unroll
        for (int r = 0; r < 8; ++r) {
            int row = row0 + mi * 16 + hk * 8 + r;
            int b = row >> 10, t = row & (TT - 1);
            float* out = xproj + ((size_t)t * BB + b) * G4 + col0 + nm;
            #pragma unroll
            for (int ni = 0; ni < 4; ++ni) out[ni * 16] = acc[mi][ni].e[r];
        }
}

// ---------------- kernel 3: persistent LSTM recurrence ----------------
// 128 waves; wave owns one 16x16 (b,j) patch and computes its i,f,g,o tiles
// (A fragment reused 4x). LSTM cell is applied fully in registers; c lives in
// VGPRs for all 1024 steps. h is double-buffered bf16 -> ONE grid barrier/step.
__global__ void k_lstm(const float* __restrict__ xproj, const __bf16* __restrict__ whhb,
                       __bf16* __restrict__ hb, __bf16* __restrict__ hsb,
                       unsigned* __restrict__ bar) {
    int wave = threadIdx.x >> 5, lane = threadIdx.x & 31;
    int w = blockIdx.x * 8 + wave;          // 0..127
    int tm = w & 3, tn = w >> 2;            // 4 b-strips x 32 j-strips
    int row0 = tm << 4, col0 = tn << 4;
    int nm = lane & 15, hk = lane >> 4;
    int gtid = blockIdx.x * 256 + threadIdx.x;   // 0..4095

    const __bf16* bpi = whhb + (size_t)(col0 + nm) * HID;
    const __bf16* bpf = whhb + (size_t)(512 + col0 + nm) * HID;
    const __bf16* bpg = whhb + (size_t)(1024 + col0 + nm) * HID;
    const __bf16* bpo = whhb + (size_t)(1536 + col0 + nm) * HID;

    float creg[8];
    #pragma unroll
    for (int r = 0; r < 8; ++r) creg[r] = 0.0f;

    unsigned tgt = 0;
    for (int t = 0; t < TT; ++t) {
        const __bf16* hread  = hb + (size_t)(t & 1) * BB * HID;
        __bf16*       hwrite = hb + (size_t)((t + 1) & 1) * BB * HID;
        const float*  xp     = xproj + (size_t)t * BB * G4;

        // warm L2 with next step's 512 KB pre-activation slab while we compute
        if (t + 1 < TT)
            __builtin_prefetch((const char*)(xproj + (size_t)(t + 1) * BB * G4) + (size_t)gtid * 128, 0, 1);

        union Acc { v8f v; float e[8]; } ai, af, ag, ao;
        #pragma unroll
        for (int r = 0; r < 8; ++r) {
            int b = row0 + hk * 8 + r;
            const float* xr = xp + (size_t)b * G4 + col0 + nm;
            ai.e[r] = xr[0];
            af.e[r] = xr[512];
            ag.e[r] = xr[1024];
            ao.e[r] = xr[1536];
        }
        const __bf16* ap = hread + (size_t)(row0 + nm) * HID;
        for (int k0 = 0; k0 < HID; k0 += 32) {
            v16bf a = load_frag16(ap + k0, hk);
            ai.v = wmma_bf16(a, load_frag16(bpi + k0, hk), ai.v);
            af.v = wmma_bf16(a, load_frag16(bpf + k0, hk), af.v);
            ag.v = wmma_bf16(a, load_frag16(bpg + k0, hk), ag.v);
            ao.v = wmma_bf16(a, load_frag16(bpo + k0, hk), ao.v);
        }

        // in-register LSTM cell for this wave's 16x16 (b,j) patch
        #pragma unroll
        for (int r = 0; r < 8; ++r) {
            float si = fast_sigmoid(ai.e[r]);
            float sf = fast_sigmoid(af.e[r]);
            float tg = fast_tanh(ag.e[r]);
            float so = fast_sigmoid(ao.e[r]);
            float cv = sf * creg[r] + si * tg;
            float hv = so * fast_tanh(cv);
            creg[r] = cv;
            int b = row0 + hk * 8 + r;
            int j = col0 + nm;
            __bf16 hvb = (__bf16)hv;
            hwrite[(size_t)b * HID + j] = hvb;
            hsb[((size_t)b * TT + t) * HID + j] = hvb;
        }

        tgt += LSTM_BLOCKS;
        grid_sync(bar, tgt);
    }
}

// ---------------- kernel 4: output projection ----------------
// logits[r, v] = hs[r,:] . W_proj[v,:] + b_proj[v]
// 32x64 strip per wave; 2048 x 64 = 131072 waves = 16384 blocks
__global__ void k_proj(const __bf16* __restrict__ hsb, const __bf16* __restrict__ wprojb,
                       const float* __restrict__ bproj, float* __restrict__ logits) {
    int wave = threadIdx.x >> 5, lane = threadIdx.x & 31;
    int w = blockIdx.x * 8 + wave;
    int tn = w & 63, tm = w >> 6;
    int row0 = tm << 5, col0 = tn << 6;
    int nm = lane & 15, hk = lane >> 4;

    const __bf16* a0 = hsb + (size_t)(row0 + nm) * HID;
    const __bf16* a1 = hsb + (size_t)(row0 + 16 + nm) * HID;
    const __bf16* bp[4];
    #pragma unroll
    for (int ni = 0; ni < 4; ++ni) bp[ni] = wprojb + (size_t)(col0 + ni * 16 + nm) * HID;

    union Acc { v8f v; float e[8]; } acc[2][4];
    #pragma unroll
    for (int ni = 0; ni < 4; ++ni) {
        float bv = bproj[col0 + ni * 16 + nm];
        #pragma unroll
        for (int mi = 0; mi < 2; ++mi)
            #pragma unroll
            for (int r = 0; r < 8; ++r) acc[mi][ni].e[r] = bv;
    }

    for (int k0 = 0; k0 < HID; k0 += 32) {
        v16bf af0 = load_frag16(a0 + k0, hk);
        v16bf af1 = load_frag16(a1 + k0, hk);
        #pragma unroll
        for (int ni = 0; ni < 4; ++ni) {
            v16bf bf = load_frag16(bp[ni] + k0, hk);
            acc[0][ni].v = wmma_bf16(af0, bf, acc[0][ni].v);
            acc[1][ni].v = wmma_bf16(af1, bf, acc[1][ni].v);
        }
    }

    #pragma unroll
    for (int mi = 0; mi < 2; ++mi)
        #pragma unroll
        for (int r = 0; r < 8; ++r) {
            float* out = logits + (size_t)(row0 + mi * 16 + hk * 8 + r) * VOCAB + col0 + nm;
            #pragma unroll
            for (int ni = 0; ni < 4; ++ni) out[ni * 16] = acc[mi][ni].e[r];
        }
}

// ---------------- kernel 5: row argmax (first-max, wave32 tree) ----------------
__global__ void k_argmax(const float* __restrict__ logits, float* __restrict__ ypred) {
    int wave = threadIdx.x >> 5, lane = threadIdx.x & 31;
    long long row = (long long)blockIdx.x * 8 + wave;  // 65536 rows
    const float* lr = logits + (size_t)row * VOCAB;

    float bv = -3.4e38f;
    int bi = 0;
    for (int j = lane; j < VOCAB; j += 32) {
        float v = lr[j];
        if (v > bv) { bv = v; bi = j; }
    }
    #pragma unroll
    for (int off = 16; off > 0; off >>= 1) {
        float ov = __shfl_xor(bv, off, 32);
        int   oi = __shfl_xor(bi, off, 32);
        if (ov > bv || (ov == bv && oi < bi)) { bv = ov; bi = oi; }
    }
    if (lane == 0) ypred[row] = (float)bi;
}

// ---------------- host launcher ----------------
extern "C" void kernel_launch(void* const* d_in, const int* in_sizes, int n_in,
                              void* d_out, int out_size, void* d_ws, size_t ws_size,
                              hipStream_t stream) {
    const int*   x      = (const int*)d_in[0];
    // d_in[1] = lens: unused by the reference computation
    const float* emb    = (const float*)d_in[2];
    const float* W_ih   = (const float*)d_in[3];
    const float* W_hh   = (const float*)d_in[4];
    const float* b_ih   = (const float*)d_in[5];
    const float* b_hh   = (const float*)d_in[6];
    const float* W_proj = (const float*)d_in[7];
    const float* b_proj = (const float*)d_in[8];

    char* ws = (char*)d_ws;
    float*    xproj  = (float*)(ws + O_XPROJ);
    __bf16*   hsb    = (__bf16*)(ws + O_HSB);
    __bf16*   embb   = (__bf16*)(ws + O_EMBB);
    __bf16*   wihb   = (__bf16*)(ws + O_WIHB);
    __bf16*   whhb   = (__bf16*)(ws + O_WHHB);
    __bf16*   wprojb = (__bf16*)(ws + O_WPRB);
    __bf16*   hb     = (__bf16*)(ws + O_HB);
    float*    bias   = (float*)(ws + O_BIAS);
    unsigned* bar    = (unsigned*)(ws + O_BAR);

    float* logits = (float*)d_out;                  // [B*T, V]
    float* ypred  = logits + (size_t)NBT * VOCAB;   // [B*T]

    k_init<<<512, 256, 0, stream>>>(emb, W_ih, W_hh, W_proj, b_ih, b_hh,
                                    embb, wihb, whhb, wprojb, bias, hb, bar);
    k_xproj<<<8192, 256, 0, stream>>>(x, embb, wihb, bias, xproj);
    k_lstm<<<LSTM_BLOCKS, 256, 0, stream>>>(xproj, whhb, hb, hsb, bar);
    k_proj<<<16384, 256, 0, stream>>>(hsb, wprojb, b_proj, logits);
    k_argmax<<<8192, 256, 0, stream>>>(logits, ypred);
}
